// FMoE_75222057222529
// MI455X (gfx1250) — matrix operand order repaired and checked by reading, not compiled, not run on hardware
//
#include <hip/hip_runtime.h>
#include <math.h>

#define N_TOKENS 4096
#define D_MODEL  1024
#define N_EXPERT 8
#define D_HIDDEN 1024
#define MAX_ROWS (N_TOKENS * 2)   // top-2: at most 8192 packed expert rows
#define KTILES   (D_MODEL / 32)

typedef __bf16 bf16;
typedef __attribute__((ext_vector_type(16))) bf16         v16bf;
typedef __attribute__((ext_vector_type(8)))  float        v8f;
typedef __attribute__((ext_vector_type(4)))  float        f32x4;
typedef __attribute__((ext_vector_type(4)))  unsigned int u32x4;
typedef __attribute__((ext_vector_type(2)))  unsigned int u32x2;

// LDS tile strides (elements): 32(+8 pad) per A row / B col -> 80 B, 16-B aligned
#define ASTRIDE 40
#define BSTRIDE 40

// ---------- helpers ----------
__device__ __forceinline__ unsigned short f32_to_bf16_bits(float f) {
  union { float f; unsigned int u; } v; v.f = f;
  unsigned int u = v.u;
  u += 0x7FFFu + ((u >> 16) & 1u);            // round-to-nearest-even
  return (unsigned short)(u >> 16);
}
__device__ __forceinline__ unsigned int pack_bf16x2(float lo, float hi) {
  return (unsigned int)f32_to_bf16_bits(lo) |
         ((unsigned int)f32_to_bf16_bits(hi) << 16);
}

union FragU { v16bf v; u32x4 q[2]; };

// ---------- 0a) weight prep: [E][K][N] f32 -> [E][N][K] bf16 (transposed) ----------
__global__ __launch_bounds__(256) void moe_wtr_kernel(
    const float* __restrict__ w, unsigned short* __restrict__ wt) {
  const int e  = blockIdx.z;
  const int kb = blockIdx.x * 32;
  const int nb = blockIdx.y * 32;
  __shared__ float tile[32][33];
  const int tid = threadIdx.x;

  const int k  = tid >> 3;
  const int n4 = (tid & 7) * 4;
  f32x4 v = *(const f32x4*)(w + ((size_t)e * 1024 + kb + k) * 1024 + nb + n4);
  tile[k][n4 + 0] = v.x; tile[k][n4 + 1] = v.y;
  tile[k][n4 + 2] = v.z; tile[k][n4 + 3] = v.w;
  __syncthreads();

  const int n  = tid >> 3;
  const int k4 = (tid & 7) * 4;
  u32x2 o;
  o.x = pack_bf16x2(tile[k4 + 0][n], tile[k4 + 1][n]);
  o.y = pack_bf16x2(tile[k4 + 2][n], tile[k4 + 3][n]);
  *(u32x2*)(wt + ((size_t)e * 1024 + nb + n) * 1024 + kb + k4) = o;
}

// ---------- 0b) activation prep: inp f32 -> bf16 (same layout) ----------
__global__ __launch_bounds__(256) void moe_xcvt_kernel(
    const float* __restrict__ inp, unsigned short* __restrict__ xb) {
  const size_t i8 = ((size_t)blockIdx.x * 256 + threadIdx.x) * 8;
  f32x4 a0 = *(const f32x4*)(inp + i8);
  f32x4 a1 = *(const f32x4*)(inp + i8 + 4);
  u32x4 o;
  o.x = pack_bf16x2(a0.x, a0.y); o.y = pack_bf16x2(a0.z, a0.w);
  o.z = pack_bf16x2(a1.x, a1.y); o.w = pack_bf16x2(a1.z, a1.w);
  *(u32x4*)(xb + i8) = o;
}

// ---------- 1) gate: logits -> top-2 -> softmax -> coeff[N,E] ----------
__global__ __launch_bounds__(256) void moe_gate_kernel(
    const float* __restrict__ inp, const float* __restrict__ gate_w,
    const float* __restrict__ gate_b, float* __restrict__ coeff) {
  const int gwave = (blockIdx.x * blockDim.x + threadIdx.x) >> 5;  // token
  const int lane  = threadIdx.x & 31;
  if (gwave >= N_TOKENS) return;
  const float* x = inp + (size_t)gwave * D_MODEL;

  float acc[N_EXPERT];
#pragma unroll
  for (int e = 0; e < N_EXPERT; ++e) acc[e] = 0.0f;

  for (int d = lane; d < D_MODEL; d += 32) {
    float xv = x[d];
#pragma unroll
    for (int e = 0; e < N_EXPERT; ++e)
      acc[e] += xv * gate_w[d * N_EXPERT + e];
  }
#pragma unroll
  for (int off = 16; off >= 1; off >>= 1)
#pragma unroll
    for (int e = 0; e < N_EXPERT; ++e)
      acc[e] += __shfl_xor(acc[e], off, 32);

  if (lane == 0) {
#pragma unroll
    for (int e = 0; e < N_EXPERT; ++e) acc[e] += gate_b[e];
    int i0 = 0;
#pragma unroll
    for (int e = 1; e < N_EXPERT; ++e) if (acc[e] > acc[i0]) i0 = e;
    int i1 = -1;
#pragma unroll
    for (int e = 0; e < N_EXPERT; ++e)
      if (e != i0 && (i1 < 0 || acc[e] > acc[i1])) i1 = e;
    float m  = fmaxf(acc[i0], acc[i1]);
    float e0 = __expf(acc[i0] - m), e1 = __expf(acc[i1] - m);
    float s  = e0 + e1;
#pragma unroll
    for (int e = 0; e < N_EXPERT; ++e) coeff[gwave * N_EXPERT + e] = 0.0f;
    coeff[gwave * N_EXPERT + i0] = e0 / s;
    coeff[gwave * N_EXPERT + i1] = e1 / s;
  }
}

// ---------- 2) compact: per-expert token lists, offsets, slots ----------
__global__ void moe_compact_kernel(const float* __restrict__ coeff,
                                   int* __restrict__ counts, int* __restrict__ offs,
                                   int* __restrict__ list, int* __restrict__ slot) {
  const int e = threadIdx.x;
  if (e < N_EXPERT) {
    int c = 0;
    for (int n = 0; n < N_TOKENS; ++n)
      if (coeff[n * N_EXPERT + e] > 0.0f) list[e * N_TOKENS + (c++)] = n;
    counts[e] = c;
  }
  __syncthreads();
  if (threadIdx.x == 0) {
    int a = 0;
    for (int i = 0; i < N_EXPERT; ++i) { offs[i] = a; a += counts[i]; }
  }
  __syncthreads();
  if (e < N_EXPERT) {
    int c = 0;
    const int base = offs[e];
    for (int n = 0; n < N_TOKENS; ++n) {
      if (coeff[n * N_EXPERT + e] > 0.0f) slot[n * N_EXPERT + e] = base + (c++);
      else                                slot[n * N_EXPERT + e] = -1;
    }
  }
}

// ---------- 3) GEMM1: h = gelu(Xbf_gathered @ W1[e] + b1[e]) -> bf16 ----------
// 64x128 block tile, BK=32, 8 waves (2x4), 2x2 wmma frags per wave.
// Double-buffered LDS (1 barrier/iter); A and B are both pre-converted bf16,
// so staging is pure b128 copies.
__global__ __launch_bounds__(256) void moe_gemm1_kernel(
    const unsigned short* __restrict__ xb, const unsigned short* __restrict__ w1t,
    const float* __restrict__ b1, const int* __restrict__ list,
    const int* __restrict__ counts, const int* __restrict__ offs,
    unsigned short* __restrict__ h) {
  const int e   = blockIdx.z;
  const int cnt = counts[e];
  const int m0  = blockIdx.x * 64;
  if (m0 >= cnt) return;                    // uniform early-exit
  const int n0  = blockIdx.y * 128;

  __shared__ unsigned short As[2][64 * ASTRIDE];    // [row][k]
  __shared__ unsigned short Bs[2][128 * BSTRIDE];   // [col][k]

  const int tid  = threadIdx.x;
  const int lane = tid & 31;
  const int wave = tid >> 5;
  const int wm   = wave & 1;
  const int wn   = wave >> 1;
  const int half = lane >> 4;
  const int lr   = lane & 15;

  // A staging: 4 threads/row, 8 consecutive k each (clamped token gather;
  // padded rows only affect padded output rows, masked at store)
  const int ar_st = tid >> 2;
  const int ak_st = (tid & 3) * 8;
  const int arow  = m0 + ar_st;
  const int tok   = list[e * N_TOKENS + (arow < cnt ? arow : cnt - 1)];
  const unsigned short* aptr = xb + (size_t)tok * D_MODEL + ak_st;

  // B staging: 2 threads/col, 16 consecutive k each (pure bf16 copy)
  const int bc_st = tid >> 1;
  const int bk_st = (tid & 1) * 16;
  const unsigned short* bptr =
      w1t + ((size_t)e * D_HIDDEN + n0 + bc_st) * D_MODEL + bk_st;

  const int aoff = ar_st * ASTRIDE + ak_st;
  const int boff = bc_st * BSTRIDE + bk_st;

  v8f acc[2][2] = {};

  // prologue: stage tile 0
  *(u32x4*)&As[0][aoff]     = *(const u32x4*)(aptr);
  *(u32x4*)&Bs[0][boff]     = *(const u32x4*)(bptr);
  *(u32x4*)&Bs[0][boff + 8] = *(const u32x4*)(bptr + 8);
  __syncthreads();

  for (int kt = 0; kt < KTILES; ++kt) {
    const int cur = kt & 1;
    const bool has_next = (kt + 1) < KTILES;

    u32x4 aN, b0N, b1N;
    if (has_next) {
      const int kn = (kt + 1) * 32;
      aN  = *(const u32x4*)(aptr + kn);
      b0N = *(const u32x4*)(bptr + kn);
      b1N = *(const u32x4*)(bptr + kn + 8);
      if (kn + 32 < D_MODEL) {                     // -> global_prefetch_b8
        __builtin_prefetch(aptr + kn + 32, 0, 1);
        __builtin_prefetch(bptr + kn + 32, 0, 1);
      }
    }

    FragU fa[2], fb[2];
#pragma unroll
    for (int mf = 0; mf < 2; ++mf) {
      const unsigned short* p = &As[cur][(wm * 32 + mf * 16 + lr) * ASTRIDE];
      fa[mf].q[0] = *(const u32x4*)(p + half * 8);        // K half*8+0..7
      fa[mf].q[1] = *(const u32x4*)(p + 16 + half * 8);   // K 16+half*8+0..7
    }
#pragma unroll
    for (int nf = 0; nf < 2; ++nf) {
      const unsigned short* p =
          &Bs[cur][(wn * 32 + nf * 16 + lr) * BSTRIDE];
      fb[nf].q[0] = *(const u32x4*)(p + half * 8);        // K half*8+0..7
      fb[nf].q[1] = *(const u32x4*)(p + 16 + half * 8);   // K 16+half*8+0..7
    }
#pragma unroll
    for (int mf = 0; mf < 2; ++mf)
#pragma unroll
      for (int nf = 0; nf < 2; ++nf)
        acc[mf][nf] = __builtin_amdgcn_wmma_f32_16x16x32_bf16(
            false, fa[mf].v, false, fb[nf].v, (short)0, acc[mf][nf],
            false, false);

    if (has_next) {
      const int nxt = cur ^ 1;
      *(u32x4*)&As[nxt][aoff]     = aN;
      *(u32x4*)&Bs[nxt][boff]     = b0N;
      *(u32x4*)&Bs[nxt][boff + 8] = b1N;
      __syncthreads();
    }
  }

  // epilogue: bias + exact-erf GeLU, store bf16 to packed h rows
  const int base = offs[e];
#pragma unroll
  for (int mf = 0; mf < 2; ++mf) {
#pragma unroll
    for (int nf = 0; nf < 2; ++nf) {
      int col = n0 + wn * 32 + nf * 16 + lr;
      float bias = b1[e * D_HIDDEN + col];
#pragma unroll
      for (int v = 0; v < 8; ++v) {          // D layout: M = v + 8*half
        int row = m0 + wm * 32 + mf * 16 + half * 8 + v;
        if (row < cnt) {
          float x = acc[mf][nf][v] + bias;
          float g = 0.5f * x * (1.0f + erff(x * 0.70710678118654752f));
          h[(size_t)(base + row) * D_HIDDEN + col] = f32_to_bf16_bits(g);
        }
      }
    }
  }
}

// ---------- 4) GEMM2: contrib = h @ W2[e] (unscaled, f32) ----------
__global__ __launch_bounds__(256) void moe_gemm2_kernel(
    const unsigned short* __restrict__ h, const unsigned short* __restrict__ w2t,
    const int* __restrict__ counts, const int* __restrict__ offs,
    float* __restrict__ contrib) {
  const int e   = blockIdx.z;
  const int cnt = counts[e];
  const int m0  = blockIdx.x * 64;
  if (m0 >= cnt) return;
  const int n0  = blockIdx.y * 128;
  const int base = offs[e];

  __shared__ unsigned short As[2][64 * ASTRIDE];
  __shared__ unsigned short Bs[2][128 * BSTRIDE];

  const int tid  = threadIdx.x;
  const int lane = tid & 31;
  const int wave = tid >> 5;
  const int wm   = wave & 1;
  const int wn   = wave >> 1;
  const int half = lane >> 4;
  const int lr   = lane & 15;

  const int ar_st = tid >> 2;
  const int ak_st = (tid & 3) * 8;
  const int arow  = m0 + ar_st;
  const int hrow  = base + (arow < cnt ? arow : cnt - 1);  // clamped, masked later
  const unsigned short* aptr = h + (size_t)hrow * D_HIDDEN + ak_st;

  const int bc_st = tid >> 1;
  const int bk_st = (tid & 1) * 16;
  const unsigned short* bptr =
      w2t + ((size_t)e * D_MODEL + n0 + bc_st) * D_HIDDEN + bk_st;

  const int aoff = ar_st * ASTRIDE + ak_st;
  const int boff = bc_st * BSTRIDE + bk_st;

  v8f acc[2][2] = {};

  *(u32x4*)&As[0][aoff]     = *(const u32x4*)(aptr);
  *(u32x4*)&Bs[0][boff]     = *(const u32x4*)(bptr);
  *(u32x4*)&Bs[0][boff + 8] = *(const u32x4*)(bptr + 8);
  __syncthreads();

  for (int kt = 0; kt < KTILES; ++kt) {
    const int cur = kt & 1;
    const bool has_next = (kt + 1) < KTILES;

    u32x4 aN, b0N, b1N;
    if (has_next) {
      const int kn = (kt + 1) * 32;
      aN  = *(const u32x4*)(aptr + kn);
      b0N = *(const u32x4*)(bptr + kn);
      b1N = *(const u32x4*)(bptr + kn + 8);
      if (kn + 32 < D_HIDDEN) {
        __builtin_prefetch(aptr + kn + 32, 0, 1);
        __builtin_prefetch(bptr + kn + 32, 0, 1);
      }
    }

    FragU fa[2], fb[2];
#pragma unroll
    for (int mf = 0; mf < 2; ++mf) {
      const unsigned short* p = &As[cur][(wm * 32 + mf * 16 + lr) * ASTRIDE];
      fa[mf].q[0] = *(const u32x4*)(p + half * 8);
      fa[mf].q[1] = *(const u32x4*)(p + 16 + half * 8);
    }
#pragma unroll
    for (int nf = 0; nf < 2; ++nf) {
      const unsigned short* p =
          &Bs[cur][(wn * 32 + nf * 16 + lr) * BSTRIDE];
      fb[nf].q[0] = *(const u32x4*)(p + half * 8);
      fb[nf].q[1] = *(const u32x4*)(p + 16 + half * 8);
    }
#pragma unroll
    for (int mf = 0; mf < 2; ++mf)
#pragma unroll
      for (int nf = 0; nf < 2; ++nf)
        acc[mf][nf] = __builtin_amdgcn_wmma_f32_16x16x32_bf16(
            false, fa[mf].v, false, fb[nf].v, (short)0, acc[mf][nf],
            false, false);

    if (has_next) {
      const int nxt = cur ^ 1;
      *(u32x4*)&As[nxt][aoff]     = aN;
      *(u32x4*)&Bs[nxt][boff]     = b0N;
      *(u32x4*)&Bs[nxt][boff + 8] = b1N;
      __syncthreads();
    }
  }

#pragma unroll
  for (int mf = 0; mf < 2; ++mf) {
#pragma unroll
    for (int nf = 0; nf < 2; ++nf) {
      int col = n0 + wn * 32 + nf * 16 + lr;
#pragma unroll
      for (int v = 0; v < 8; ++v) {
        int row = m0 + wm * 32 + mf * 16 + half * 8 + v;
        if (row < cnt)
          contrib[(size_t)(base + row) * D_MODEL + col] = acc[mf][nf][v];
      }
    }
  }
}

// ---------- 5) combine: out[n,:] = sum_e coeff * (b2[e,:] + contrib[slot]) ----------
__global__ __launch_bounds__(256) void moe_combine_kernel(
    const float* __restrict__ coeff, const float* __restrict__ b2,
    const int* __restrict__ slot, const float* __restrict__ contrib,
    float* __restrict__ out) {
  int idx = blockIdx.x * 256 + threadIdx.x;
  if (idx >= N_TOKENS * D_MODEL) return;
  int n = idx >> 10;
  int d = idx & (D_MODEL - 1);
  float s = 0.0f;
#pragma unroll
  for (int e = 0; e < N_EXPERT; ++e) {
    float c = coeff[n * N_EXPERT + e];
    if (c > 0.0f) {
      int sl = slot[n * N_EXPERT + e];
      s += c * (b2[e * D_MODEL + d] + contrib[(size_t)sl * D_MODEL + d]);
    }
  }
  out[idx] = s;
}

// ---------- launch ----------
extern "C" void kernel_launch(void* const* d_in, const int* in_sizes, int n_in,
                              void* d_out, int out_size, void* d_ws, size_t ws_size,
                              hipStream_t stream) {
  (void)in_sizes; (void)n_in; (void)out_size; (void)ws_size;
  const float* inp    = (const float*)d_in[0];
  const float* gate_w = (const float*)d_in[1];
  const float* gate_b = (const float*)d_in[2];
  const float* w1     = (const float*)d_in[3];
  const float* b1     = (const float*)d_in[4];
  const float* w2     = (const float*)d_in[5];
  const float* b2     = (const float*)d_in[6];
  float* out = (float*)d_out;

  char* p = (char*)d_ws;
  auto take = [&](size_t bytes) {
    char* q = p; p += (bytes + 255) & ~(size_t)255; return q;
  };
  float*          coeff   = (float*)take((size_t)N_TOKENS * N_EXPERT * 4);
  int*            counts  = (int*)take(N_EXPERT * 4);
  int*            offs    = (int*)take(N_EXPERT * 4);
  int*            list    = (int*)take((size_t)N_EXPERT * N_TOKENS * 4);
  int*            slot    = (int*)take((size_t)N_TOKENS * N_EXPERT * 4);
  unsigned short* h       = (unsigned short*)take((size_t)MAX_ROWS * D_HIDDEN * 2);
  float*          contrib = (float*)take((size_t)MAX_ROWS * D_MODEL * 4);
  unsigned short* w1t     = (unsigned short*)take((size_t)N_EXPERT * D_MODEL * D_HIDDEN * 2);
  unsigned short* w2t     = (unsigned short*)take((size_t)N_EXPERT * D_HIDDEN * D_MODEL * 2);
  unsigned short* xb      = (unsigned short*)take((size_t)N_TOKENS * D_MODEL * 2);

  dim3 gt(32, 32, N_EXPERT);
  moe_wtr_kernel<<<gt, 256, 0, stream>>>(w1, w1t);   // bf16 + transpose
  moe_wtr_kernel<<<gt, 256, 0, stream>>>(w2, w2t);
  moe_xcvt_kernel<<<(N_TOKENS * D_MODEL) / (256 * 8), 256, 0, stream>>>(inp, xb);

  moe_gate_kernel<<<N_TOKENS / 8, 256, 0, stream>>>(inp, gate_w, gate_b, coeff);
  moe_compact_kernel<<<1, 32, 0, stream>>>(coeff, counts, offs, list, slot);

  dim3 g1(N_TOKENS / 64, D_HIDDEN / 128, N_EXPERT);
  moe_gemm1_kernel<<<g1, 256, 0, stream>>>(xb, w1t, b1, list, counts, offs, h);

  dim3 g2(N_TOKENS / 64, D_MODEL / 128, N_EXPERT);
  moe_gemm2_kernel<<<g2, 256, 0, stream>>>(h, w2t, counts, offs, contrib);

  moe_combine_kernel<<<(N_TOKENS * D_MODEL) / 256, 256, 0, stream>>>(
      coeff, b2, slot, contrib, out);
}
